// StepModel_12506944766390
// MI455X (gfx1250) — matrix operand order repaired and checked by reading, not compiled, not run on hardware
//
#include <hip/hip_runtime.h>
#include <math.h>

// ---------------------------------------------------------------------------
// CDNA5 (gfx1250) implementation of the point-transformer step model.
// All 1x1 convs run through v_wmma_f32_16x16x32_f16 (f16 in, f32 accum).
// Activations are point-major X[col][Kpad32]; weights W[Mpad16][Kpad32].
// ---------------------------------------------------------------------------

typedef _Float16 v8h  __attribute__((ext_vector_type(8)));
typedef _Float16 v16h __attribute__((ext_vector_type(16)));
typedef float    v8f  __attribute__((ext_vector_type(8)));

#define DEV_INLINE __device__ __forceinline__

static inline size_t cdiv_sz(size_t a, size_t b) { return (a + b - 1) / b; }

// ------------------------------- GEMM --------------------------------------
// Y[col][M] = sum_k W[m][k] * X[col][k] + bias[m]
// W: [Mpad][Kpad] f16, X: [Ncols][Kpad] f16, Ncols multiple of 16.
__global__ __launch_bounds__(256) void k_gemm(
    const _Float16* __restrict__ W, const _Float16* __restrict__ X,
    const float* __restrict__ bias, float* __restrict__ Y,
    int M, int Kpad, int Ncols) {
  int lane = threadIdx.x & 31;
  int wave = threadIdx.x >> 5;
  int colTile = blockIdx.x * 8 + wave;
  if (colTile * 16 >= Ncols) return;          // wave-uniform guard (EXEC stays full)
  int m0 = blockIdx.y * 16;
  int r  = lane & 15;
  int hi = lane >> 4;
  const _Float16* wrow = W + (size_t)(m0 + r) * Kpad + 16 * hi;
  const _Float16* xrow = X + (size_t)(colTile * 16 + r) * Kpad + 16 * hi;
  v8f acc = {};
  for (int k0 = 0; k0 < Kpad; k0 += 32) {
    v8h  alo = *(const v8h*)(wrow + k0);
    v8h  ahi = *(const v8h*)(wrow + k0 + 8);
    v16h a = __builtin_shufflevector(alo, ahi, 0,1,2,3,4,5,6,7,8,9,10,11,12,13,14,15);
    v16h b = *(const v16h*)(xrow + k0);
    acc = __builtin_amdgcn_wmma_f32_16x16x32_f16(false, a, false, b, (short)0, acc,
                                                 false, false);
  }
  int mb = m0 + 8 * hi;
  size_t base = (size_t)(colTile * 16 + r) * M;
#pragma unroll
  for (int v = 0; v < 8; ++v) {
    int m = mb + v;
    if (m < M) Y[base + m] = acc[v] + bias[m];
  }
}

// --------------------------- weight convert --------------------------------
__global__ void k_convert_w(const float* __restrict__ w, int M, int K, int Kpad,
                            int Mpad, _Float16* __restrict__ o) {
  int e = blockIdx.x * blockDim.x + threadIdx.x;
  if (e >= Mpad * Kpad) return;
  int m = e / Kpad, k = e % Kpad;
  o[e] = (m < M && k < K) ? (_Float16)w[(size_t)m * K + k] : (_Float16)0.f;
}

// ------------------------------ BN stats -----------------------------------
__global__ __launch_bounds__(256) void k_bn_stats(const float* __restrict__ Y,
                                                  int M, int Ncols,
                                                  float* mean, float* rstd) {
  int c = blockIdx.x, tid = threadIdx.x;
  float s = 0.f, s2 = 0.f;
  for (int col = tid; col < Ncols; col += 256) {
    float v = Y[(size_t)col * M + c];
    s += v; s2 += v * v;
  }
  __shared__ float sh[256], sh2[256];
  sh[tid] = s; sh2[tid] = s2;
  __syncthreads();
  for (int o = 128; o > 0; o >>= 1) {
    if (tid < o) { sh[tid] += sh[tid + o]; sh2[tid] += sh2[tid + o]; }
    __syncthreads();
  }
  if (tid == 0) {
    float mn = sh[0] / (float)Ncols;
    float var = sh2[0] / (float)Ncols - mn * mn;
    mean[c] = mn;
    rstd[c] = rsqrtf(var + 1e-5f);
  }
}

// ----------------------- BN / ReLU / pack-to-f16 ---------------------------
__global__ void k_bn_act(const float* __restrict__ Y, int M, int Ncols,
                         const float* mean, const float* rstd,
                         const float* gam, const float* bet, int relu,
                         _Float16* O16, int Kpad16, float* O32, int J) {
  size_t e = (size_t)blockIdx.x * blockDim.x + threadIdx.x;
  size_t tot = (size_t)Ncols * J;
  if (e >= tot) return;
  int col = (int)(e / J), j = (int)(e % J);
  float v = 0.f;
  if (j < M) {
    v = Y[(size_t)col * M + j];
    if (mean) v = (v - mean[j]) * rstd[j] * gam[j] + bet[j];
    if (relu) v = fmaxf(v, 0.f);
    if (O32) O32[(size_t)col * M + j] = v;
  }
  if (O16) O16[(size_t)col * Kpad16 + j] = (_Float16)v;
}

// ------------------------------ max pool -----------------------------------
__global__ void k_maxpool(const float* __restrict__ Yin, int C, int K,
                          int NcolsOut, float* O32, _Float16* O16, int Kpad16,
                          int J) {
  size_t e = (size_t)blockIdx.x * blockDim.x + threadIdx.x;
  size_t tot = (size_t)NcolsOut * J;
  if (e >= tot) return;
  int colS = (int)(e / J), j = (int)(e % J);
  float v = 0.f;
  if (j < C) {
    float m = -3.4e38f;
    const float* p = Yin + (size_t)colS * K * C + j;
    for (int k = 0; k < K; ++k) m = fmaxf(m, p[(size_t)k * C]);
    v = m;
    if (O32) O32[(size_t)colS * C + j] = m;
  }
  if (O16) O16[(size_t)colS * Kpad16 + j] = (_Float16)v;
}

// -------------------------------- FPS --------------------------------------
__global__ __launch_bounds__(256) void k_fps(const float* __restrict__ xyz,
                                             int N, int S, int* outIdx) {
  int b = blockIdx.x, tid = threadIdx.x;
  const int NT = 256;
  __shared__ float dist[2048];
  __shared__ float rv[NT];
  __shared__ int   ri[NT];
  __shared__ int   sFar;
  for (int j = tid; j < N; j += NT) dist[j] = 1e10f;
  if (tid == 0) sFar = 0;
  __syncthreads();
  const float* xb = xyz + (size_t)b * 3 * N;
  for (int it = 0; it < S; ++it) {
    int far = sFar;
    if (tid == 0) outIdx[b * S + it] = far;
    float cx = xb[far], cy = xb[N + far], cz = xb[2 * N + far];
    float best = -1.f; int bi = 0;
    for (int j = tid; j < N; j += NT) {
      float dx = xb[j] - cx, dy = xb[N + j] - cy, dz = xb[2 * N + j] - cz;
      float d = dx * dx + dy * dy + dz * dz;
      float nd = fminf(dist[j], d);
      dist[j] = nd;
      if (nd > best) { best = nd; bi = j; }
    }
    rv[tid] = best; ri[tid] = bi;
    __syncthreads();
    for (int o = NT / 2; o > 0; o >>= 1) {
      if (tid < o) {
        if (rv[tid + o] > rv[tid] ||
            (rv[tid + o] == rv[tid] && ri[tid + o] < ri[tid])) {
          rv[tid] = rv[tid + o]; ri[tid] = ri[tid + o];
        }
      }
      __syncthreads();
    }
    if (tid == 0) sFar = ri[0];
    __syncthreads();
  }
}

// ---------------------------- gather xyz -----------------------------------
__global__ void k_gather_xyz(const float* __restrict__ xyz, int N,
                             const int* __restrict__ idx, int S, float* out) {
  int e = blockIdx.x * blockDim.x + threadIdx.x;
  if (e >= 16 * 3 * S) return;
  int b = e / (3 * S), r = e % (3 * S), c = r / S, s = r % S;
  out[e] = xyz[((size_t)b * 3 + c) * N + idx[b * S + s]];
}

// ---------------------------- ball query -----------------------------------
__global__ void k_ball_query(const float* __restrict__ xyz,
                             const float* __restrict__ nxyz, int N, int S,
                             float r2, int* __restrict__ out) {
  int g = blockIdx.x * blockDim.x + threadIdx.x;
  if (g >= 16 * S) return;
  int b = g / S, s = g % S;
  const float* xb = xyz + (size_t)b * 3 * N;
  const float* nb = nxyz + (size_t)b * 3 * S;
  float cx = nb[s], cy = nb[S + s], cz = nb[2 * S + s];
  float sc = cx * cx + cy * cy + cz * cz;
  int cnt = 0, first = 0; bool have = false;
  int* o = out + (size_t)g * 32;
  for (int j = 0; j < N && cnt < 32; ++j) {
    float px = xb[j], py = xb[N + j], pz = xb[2 * N + j];
    float d = sc + px * px + py * py + pz * pz - 2.f * (cx * px + cy * py + cz * pz);
    if (!(d > r2)) {
      if (!have) { first = j; have = true; }
      o[cnt++] = j;
    }
  }
  for (; cnt < 32; ++cnt) o[cnt] = first;
}

// ------------------------------ grouping -----------------------------------
__global__ void k_group(const float* __restrict__ xyz,
                        const float* __restrict__ nxyz,
                        const int* __restrict__ idx,
                        const float* __restrict__ pts, int pts_is_xyz, int N,
                        int S, int K, int Cp, int Kpad, _Float16* O) {
  size_t e = (size_t)blockIdx.x * blockDim.x + threadIdx.x;
  size_t tot = (size_t)16 * S * K * Kpad;
  if (e >= tot) return;
  int col2 = (int)(e / Kpad), c = (int)(e % Kpad);
  int b = col2 / (S * K), rem = col2 % (S * K), s = rem / K;
  int j = idx[col2];
  float v = 0.f;
  if (c < 3) {
    v = xyz[((size_t)b * 3 + c) * N + j] - nxyz[((size_t)b * 3 + c) * S + s];
  } else if (c < 3 + Cp) {
    v = pts_is_xyz ? xyz[((size_t)b * 3 + (c - 3)) * N + j]
                   : pts[(size_t)(b * N + j) * Cp + (c - 3)];
  }
  O[e] = (_Float16)v;
}

__global__ void k_group_all(const float* __restrict__ x2,
                            const float* __restrict__ pts, int n, int Cp,
                            int Kpad, _Float16* O) {
  size_t e = (size_t)blockIdx.x * blockDim.x + threadIdx.x;
  size_t tot = (size_t)16 * n * Kpad;
  if (e >= tot) return;
  int col = (int)(e / Kpad), c = (int)(e % Kpad);
  int b = col / n, kk = col % n;
  float v = 0.f;
  if (c < 3) v = x2[((size_t)b * 3 + c) * n + kk];
  else if (c < 3 + Cp) v = pts[(size_t)col * Cp + (c - 3)];
  O[e] = (_Float16)v;
}

// -------------------------------- kNN --------------------------------------
__global__ void k_knn(const float* __restrict__ xyz, int n, int* __restrict__ out) {
  int g = blockIdx.x * blockDim.x + threadIdx.x;
  if (g >= 16 * n) return;
  int b = g / n, i = g % n;
  const float* xb = xyz + (size_t)b * 3 * n;
  float cx = xb[i], cy = xb[n + i], cz = xb[2 * n + i];
  float sc = cx * cx + cy * cy + cz * cz;
  float bd[16]; int bi[16];
  for (int t = 0; t < 16; ++t) { bd[t] = 3.4e38f; bi[t] = 0; }
  for (int j = 0; j < n; ++j) {
    float px = xb[j], py = xb[n + j], pz = xb[2 * n + j];
    float d = sc + px * px + py * py + pz * pz - 2.f * (cx * px + cy * py + cz * pz);
    if (d < bd[15]) {
      int p = 15;
      while (p > 0 && d < bd[p - 1]) { bd[p] = bd[p - 1]; bi[p] = bi[p - 1]; --p; }
      bd[p] = d; bi[p] = j;
    }
  }
  for (int t = 0; t < 16; ++t) out[(size_t)g * 16 + t] = bi[t];
}

// -------------------------- transformer helpers ----------------------------
__global__ void k_pos_rel(const float* __restrict__ pos,
                          const int* __restrict__ knn, int n, _Float16* O) {
  size_t e = (size_t)blockIdx.x * blockDim.x + threadIdx.x;
  size_t tot = (size_t)16 * n * 16 * 32;
  if (e >= tot) return;
  int col2 = (int)(e / 32), c = (int)(e % 32);
  int col = col2 >> 4, kk = col2 & 15;
  int b = col / n, i = col % n;
  float v = 0.f;
  if (c < 3) {
    int j = knn[(size_t)col * 16 + kk];
    v = pos[((size_t)b * 3 + c) * n + i] - pos[((size_t)b * 3 + c) * n + j];
  }
  O[e] = (_Float16)v;
}

__global__ void k_attn_in(const float* __restrict__ q,
                          const float* __restrict__ kb,
                          const float* __restrict__ pe,
                          const int* __restrict__ knn, int n, _Float16* O) {
  size_t e = (size_t)blockIdx.x * blockDim.x + threadIdx.x;
  size_t tot = (size_t)16 * n * 16 * 64;
  if (e >= tot) return;
  int col2 = (int)(e / 64), c = (int)(e % 64);
  int col = col2 >> 4, kk = col2 & 15;
  int b = col / n;
  int j = knn[(size_t)col * 16 + kk];
  float v = q[(size_t)col * 64 + c] - kb[(size_t)(b * n + j) * 64 + c] +
            pe[(size_t)col2 * 64 + c];
  O[e] = (_Float16)v;
}

__global__ void k_softmax(float* __restrict__ Y, int n) {
  size_t e = (size_t)blockIdx.x * blockDim.x + threadIdx.x;
  size_t tot = (size_t)16 * n * 64;
  if (e >= tot) return;
  int col = (int)(e / 64), c = (int)(e % 64);
  float* p = Y + ((size_t)col * 16) * 64 + c;
  float mx = p[0];
  for (int k = 1; k < 16; ++k) mx = fmaxf(mx, p[(size_t)k * 64]);
  float ex[16]; float s = 0.f;
  for (int k = 0; k < 16; ++k) { ex[k] = expf(p[(size_t)k * 64] - mx); s += ex[k]; }
  for (int k = 0; k < 16; ++k) p[(size_t)k * 64] = ex[k] / s;
}

__global__ void k_agg(const float* __restrict__ attn,
                      const float* __restrict__ v,
                      const float* __restrict__ pe, int n, _Float16* O) {
  size_t e = (size_t)blockIdx.x * blockDim.x + threadIdx.x;
  size_t tot = (size_t)16 * n * 64;
  if (e >= tot) return;
  int col = (int)(e / 64), c = (int)(e % 64);
  float v0 = v[(size_t)col * 64 + c];
  float acc = 0.f;
  for (int k = 0; k < 16; ++k) {
    size_t i2 = (size_t)(col * 16 + k) * 64 + c;
    acc += attn[i2] * (v0 + pe[i2]);
  }
  O[(size_t)col * 64 + c] = (_Float16)acc;
}

__global__ void k_add_id(const float* __restrict__ Y,
                         const float* __restrict__ idn, int C, int Ncols,
                         float* O32, _Float16* O16) {
  size_t e = (size_t)blockIdx.x * blockDim.x + threadIdx.x;
  size_t tot = (size_t)Ncols * C;
  if (e >= tot) return;
  float v = Y[e] + idn[e];
  O32[e] = v;
  O16[e] = (_Float16)v;
}

// -------------------------- fp-module helpers ------------------------------
__global__ void k_three_nn(const float* __restrict__ x1, int N,
                           const float* __restrict__ x2, int m,
                           int* __restrict__ idx3, float* __restrict__ w3) {
  int g = blockIdx.x * blockDim.x + threadIdx.x;
  if (g >= 16 * N) return;
  if (m == 1) {
    idx3[g * 3] = 0; w3[g * 3] = 1.f;
    idx3[g * 3 + 1] = 0; w3[g * 3 + 1] = 0.f;
    idx3[g * 3 + 2] = 0; w3[g * 3 + 2] = 0.f;
    return;
  }
  int b = g / N, i = g % N;
  const float* xb = x1 + (size_t)b * 3 * N;
  const float* yb = x2 + (size_t)b * 3 * m;
  float cx = xb[i], cy = xb[N + i], cz = xb[2 * N + i];
  float sc = cx * cx + cy * cy + cz * cz;
  float bd[3] = {1e30f, 1e30f, 1e30f};
  int bi[3] = {0, 0, 0};
  for (int j = 0; j < m; ++j) {
    float px = yb[j], py = yb[m + j], pz = yb[2 * m + j];
    float d = sc + px * px + py * py + pz * pz - 2.f * (cx * px + cy * py + cz * pz);
    if (d < bd[2]) {
      int p = 2;
      while (p > 0 && d < bd[p - 1]) { bd[p] = bd[p - 1]; bi[p] = bi[p - 1]; --p; }
      bd[p] = d; bi[p] = j;
    }
  }
  int kk = m < 3 ? m : 3;
  float r[3], s = 0.f;
  for (int t = 0; t < kk; ++t) {
    float dd = fmaxf(bd[t], 1e-10f);
    r[t] = 1.f / dd; s += r[t];
  }
  for (int t = 0; t < 3; ++t) {
    if (t < kk) { idx3[g * 3 + t] = bi[t]; w3[g * 3 + t] = r[t] / s; }
    else { idx3[g * 3 + t] = 0; w3[g * 3 + t] = 0.f; }
  }
}

__global__ void k_interp_concat(const float* __restrict__ p2, int C2, int m,
                                const float* __restrict__ p1, int C1,
                                const int* __restrict__ idx3,
                                const float* __restrict__ w3, int N, int Kpad,
                                _Float16* O) {
  size_t e = (size_t)blockIdx.x * blockDim.x + threadIdx.x;
  size_t tot = (size_t)16 * N * Kpad;
  if (e >= tot) return;
  int col = (int)(e / Kpad), c = (int)(e % Kpad);
  int b = col / N;
  float v = 0.f;
  if (c < C2) {
    for (int t = 0; t < 3; ++t) {
      float wv = w3[(size_t)col * 3 + t];
      if (wv != 0.f) {
        int jd = idx3[(size_t)col * 3 + t];
        v += wv * p2[(size_t)(b * m + jd) * C2 + c];
      }
    }
  } else if (c < C2 + C1) {
    v = p1[(size_t)col * C1 + (c - C2)];
  }
  O[e] = (_Float16)v;
}

// --------------------------- misc small kernels ----------------------------
__global__ void k_build_pc6(const float* __restrict__ pc, float* __restrict__ o) {
  int e = blockIdx.x * blockDim.x + threadIdx.x;
  if (e >= 16 * 2048 * 6) return;
  int col = e / 6, c = e % 6;
  int b = col / 2048, i = col % 2048;
  o[e] = pc[((size_t)b * 3 + (c % 3)) * 2048 + i];
}

__global__ void k_lat(const float* __restrict__ l3p, const float* __restrict__ fl,
                      const float* __restrict__ w, const float* __restrict__ bb,
                      float* __restrict__ y) {
  int t = blockIdx.x * blockDim.x + threadIdx.x;
  if (t >= 16 * 1024) return;
  y[t] = w[0] * l3p[t] + w[1] * fl[t] + bb[0];
}

__global__ void k_concat_noise(const float* __restrict__ l0,
                               const float* __restrict__ noise,
                               _Float16* __restrict__ O) {
  size_t e = (size_t)blockIdx.x * blockDim.x + threadIdx.x;
  size_t tot = (size_t)16 * 2048 * 160;
  if (e >= tot) return;
  int col = (int)(e / 160), c = (int)(e % 160);
  int b = col / 2048, i = col % 2048;
  float v = (c < 128) ? l0[(size_t)col * 128 + c]
                      : noise[((size_t)b * 32 + (c - 128)) * 2048 + i];
  O[e] = (_Float16)v;
}

__global__ void k_final_out(const float* __restrict__ y3,
                            const float* __restrict__ pc, float* __restrict__ o2) {
  int e = blockIdx.x * blockDim.x + threadIdx.x;
  if (e >= 16 * 3 * 2048) return;
  int b = e / (3 * 2048), r = e % (3 * 2048), c = r / 2048, i = r % 2048;
  o2[e] = pc[e] + tanhf(y3[(size_t)(b * 2048 + i) * 3 + c]);
}

// ---------------------------------------------------------------------------
extern "C" void kernel_launch(void* const* d_in, const int* in_sizes, int n_in,
                              void* d_out, int out_size, void* d_ws, size_t ws_size,
                              hipStream_t stream) {
  (void)in_sizes; (void)n_in; (void)out_size; (void)ws_size;
  const int B = 16, N0 = 2048, N1 = 512, N2 = 128;
  const int NK1 = B * N1 * 32;     // 262144
  const int NK2 = B * N2 * 32;     // 65536
  const int CC1 = B * N1;          // 8192
  const int CC2 = B * N2;          // 2048
  const int CN0 = B * N0;          // 32768
  const int CK1 = CC1 * 16;        // 131072
  const int CK2 = CC2 * 16;        // 32768

  const float* pc    = (const float*)d_in[0];
  const float* Flast = (const float*)d_in[1];
  const float* noise = (const float*)d_in[2];

  // ---- params: JAX pytree flatten (dicts in sorted-key order) ----
  int pi = 3;
  auto nf = [&]() -> const float* { return (const float*)d_in[pi++]; };
  struct CP { const float *b, *beta, *g, *w; };
  auto convbn = [&]() { CP c; c.b = nf(); c.beta = nf(); c.g = nf(); c.w = nf(); return c; };
  auto convnb = [&]() { CP c; c.b = nf(); c.beta = nullptr; c.g = nullptr; c.w = nf(); return c; };

  const float* bn1_beta = nf();
  const float* bn1_g    = nf();
  CP conv1 = convnb();
  CP fin0 = convbn(), fin1 = convbn(), fin2 = convnb();
  CP fp1m0 = convbn(), fp1m1 = convbn(), fp1m2 = convbn();
  CP fp2m0 = convbn(), fp2m1 = convbn();
  CP fp3m0 = convbn(), fp3m1 = convbn();
  CP sa1m0 = convbn(), sa1m1 = convbn(), sa1m2 = convbn();
  CP sa2m0 = convbn(), sa2m1 = convbn(), sa2m2 = convbn();
  CP sa3m0 = convbn(), sa3m1 = convbn(), sa3m2 = convbn();
  struct TR { CP attn1, attn2; const float *attn_beta, *attn_g;
              CP key, lin_end, lin_start, pos1, pos2;
              const float *pos_beta, *pos_g; CP query, value; };
  auto tr_read = [&]() { TR t;
    t.attn1 = convnb(); t.attn2 = convnb();
    t.attn_beta = nf(); t.attn_g = nf();
    t.key = convnb(); t.lin_end = convnb(); t.lin_start = convnb();
    t.pos1 = convnb(); t.pos2 = convnb();
    t.pos_beta = nf(); t.pos_g = nf();
    t.query = convnb(); t.value = convnb();
    return t; };
  TR tr1 = tr_read(), tr2 = tr_read();

  // ---- workspace layout ----
  size_t off = 0;
  auto alloc = [&](size_t bytes) -> void* {
    off = (off + 255) & ~(size_t)255;
    void* p = (char*)d_ws + off;
    off += bytes;
    return p;
  };
  float*    Y0   = (float*)alloc((size_t)33554432 * 4);
  float*    Y1   = (float*)alloc((size_t)33554432 * 4);
  _Float16* H0   = (_Float16*)alloc((size_t)33554432 * 2);
  _Float16* H1   = (_Float16*)alloc((size_t)16777216 * 2);
  _Float16* WBUF = (_Float16*)alloc((size_t)2097152 * 2);
  float* TposEmb = (float*)alloc((size_t)8388608 * 4);
  float* Tq = (float*)alloc((size_t)524288 * 4);
  float* Tk = (float*)alloc((size_t)524288 * 4);
  float* Tv = (float*)alloc((size_t)524288 * 4);
  _Float16* Txs16  = (_Float16*)alloc((size_t)524288 * 2);
  _Float16* Tagg16 = (_Float16*)alloc((size_t)524288 * 2);
  float* l1xyz = (float*)alloc((size_t)B * 3 * N1 * 4);
  float* l2xyz = (float*)alloc((size_t)B * 3 * N2 * 4);
  float*    l1A32 = (float*)alloc((size_t)CC1 * 128 * 4);
  _Float16* l1A16 = (_Float16*)alloc((size_t)CC1 * 128 * 2);
  float*    l1B32 = (float*)alloc((size_t)CC1 * 128 * 4);
  _Float16* l1B16 = (_Float16*)alloc((size_t)CC1 * 128 * 2);
  float*    l1C32 = (float*)alloc((size_t)CC1 * 128 * 4);
  float*    l2A32 = (float*)alloc((size_t)CC2 * 256 * 4);
  _Float16* l2A16 = (_Float16*)alloc((size_t)CC2 * 256 * 2);
  float*    l2B32 = (float*)alloc((size_t)CC2 * 256 * 4);
  _Float16* l2B16 = (_Float16*)alloc((size_t)CC2 * 256 * 2);
  float*    l2C32 = (float*)alloc((size_t)CC2 * 256 * 4);
  float* l3pA  = (float*)alloc((size_t)B * 1024 * 4);
  float* latY  = (float*)alloc((size_t)B * 1024 * 4);
  float* l0pts = (float*)alloc((size_t)CN0 * 128 * 4);
  float* pc6   = (float*)alloc((size_t)CN0 * 6 * 4);
  int* fpsIdx1 = (int*)alloc((size_t)B * N1 * 4);
  int* fpsIdx2 = (int*)alloc((size_t)B * N2 * 4);
  int* bq1  = (int*)alloc((size_t)NK1 * 4);
  int* bq2  = (int*)alloc((size_t)NK2 * 4);
  int* knn1 = (int*)alloc((size_t)CK1 * 4);
  int* knn2 = (int*)alloc((size_t)CK2 * 4);
  int*   idx3 = (int*)alloc((size_t)CN0 * 3 * 4);
  float* w3   = (float*)alloc((size_t)CN0 * 3 * 4);
  float* dmean = (float*)alloc(1024 * 4);
  float* drstd = (float*)alloc(1024 * 4);

  // ---- weight prep (f16, zero-padded to [Mpad16][Kpad32]) ----
  struct GW { _Float16* w; int M; int Kpad; const float* bias; };
  size_t woff = 0;
  auto prepw = [&](const CP& c, int M, int K) -> GW {
    int Mpad = (M + 15) & ~15, Kpad = (K + 31) & ~31;
    _Float16* dst = WBUF + woff;
    woff += (size_t)Mpad * Kpad;
    int tot = Mpad * Kpad;
    k_convert_w<<<(int)cdiv_sz(tot, 256), 256, 0, stream>>>(c.w, M, K, Kpad, Mpad, dst);
    return GW{dst, M, Kpad, c.b};
  };
  GW g_sa1_0 = prepw(sa1m0, 64, 6),   g_sa1_1 = prepw(sa1m1, 64, 64),  g_sa1_2 = prepw(sa1m2, 128, 64);
  GW g_sa2_0 = prepw(sa2m0, 128, 131),g_sa2_1 = prepw(sa2m1, 128, 128),g_sa2_2 = prepw(sa2m2, 256, 128);
  GW g_sa3_0 = prepw(sa3m0, 256, 259),g_sa3_1 = prepw(sa3m1, 512, 256),g_sa3_2 = prepw(sa3m2, 1024, 512);
  GW g_fp3_0 = prepw(fp3m0, 256, 1280), g_fp3_1 = prepw(fp3m1, 256, 256);
  GW g_fp2_0 = prepw(fp2m0, 256, 384),  g_fp2_1 = prepw(fp2m1, 128, 256);
  GW g_fp1_0 = prepw(fp1m0, 128, 134),  g_fp1_1 = prepw(fp1m1, 128, 128), g_fp1_2 = prepw(fp1m2, 128, 128);
  GW g_fin0 = prepw(fin0, 128, 160), g_fin1 = prepw(fin1, 64, 128), g_fin2 = prepw(fin2, 3, 64);
  struct TGW { GW ls, k, q, v, p1, p2, a1, a2, le; };
  auto prep_tr = [&](const TR& t, int inch) -> TGW {
    TGW o;
    o.ls = prepw(t.lin_start, 64, inch);
    o.k  = prepw(t.key, 64, 64);
    o.q  = prepw(t.query, 64, 64);
    o.v  = prepw(t.value, 64, 64);
    o.p1 = prepw(t.pos1, 64, 3);
    o.p2 = prepw(t.pos2, 64, 64);
    o.a1 = prepw(t.attn1, 256, 64);
    o.a2 = prepw(t.attn2, 64, 256);
    o.le = prepw(t.lin_end, inch, 64);
    return o;
  };
  TGW gtr1 = prep_tr(tr1, 128), gtr2 = prep_tr(tr2, 256);

  // ---- common wrappers ----
  auto gemm = [&](const GW& gw, const _Float16* X, int Ncols, float* Y) {
    dim3 grid((unsigned)cdiv_sz(Ncols, 128), (unsigned)cdiv_sz(gw.M, 16));
    k_gemm<<<grid, 256, 0, stream>>>(gw.w, X, gw.bias, Y, gw.M, gw.Kpad, Ncols);
  };
  auto stats = [&](const float* Y, int M, int Ncols) {
    k_bn_stats<<<M, 256, 0, stream>>>(Y, M, Ncols, dmean, drstd);
  };
  auto bnact = [&](const float* Y, int M, int Ncols, const float* mn,
                   const float* rs, const float* gg, const float* be, int relu,
                   _Float16* O16, int Kpad16, float* O32) {
    int J = O16 ? Kpad16 : M;
    size_t tot = (size_t)Ncols * J;
    k_bn_act<<<(int)cdiv_sz(tot, 256), 256, 0, stream>>>(Y, M, Ncols, mn, rs, gg,
                                                         be, relu, O16, Kpad16, O32, J);
  };
  auto maxpool = [&](const float* Yin, int C, int K, int NcolsOut, float* O32,
                     _Float16* O16, int Kpad16) {
    int J = O16 ? Kpad16 : C;
    size_t tot = (size_t)NcolsOut * J;
    k_maxpool<<<(int)cdiv_sz(tot, 256), 256, 0, stream>>>(Yin, C, K, NcolsOut,
                                                          O32, O16, Kpad16, J);
  };
  auto run_tr = [&](const TGW& g, const TR& t, int n, int inch,
                    const float* xyzcm, const float* A32, const _Float16* A16,
                    float* B32, _Float16* B16, int* knnbuf) {
    int cols = B * n, colsk = cols * 16;
    k_knn<<<(int)cdiv_sz(cols, 128), 128, 0, stream>>>(xyzcm, n, knnbuf);
    gemm(g.ls, A16, cols, Y0);
    bnact(Y0, 64, cols, nullptr, nullptr, nullptr, nullptr, 0, Txs16, 64, nullptr);
    gemm(g.k, Txs16, cols, Tk);
    gemm(g.q, Txs16, cols, Tq);
    gemm(g.v, Txs16, cols, Tv);
    k_pos_rel<<<(int)cdiv_sz((size_t)colsk * 32, 256), 256, 0, stream>>>(xyzcm, knnbuf, n, H1);
    gemm(g.p1, H1, colsk, Y0);
    stats(Y0, 64, colsk);
    bnact(Y0, 64, colsk, dmean, drstd, t.pos_g, t.pos_beta, 1, H1, 64, nullptr);
    gemm(g.p2, H1, colsk, TposEmb);
    k_attn_in<<<(int)cdiv_sz((size_t)colsk * 64, 256), 256, 0, stream>>>(Tq, Tk, TposEmb, knnbuf, n, H1);
    gemm(g.a1, H1, colsk, Y0);
    stats(Y0, 256, colsk);
    bnact(Y0, 256, colsk, dmean, drstd, t.attn_g, t.attn_beta, 1, H0, 256, nullptr);
    gemm(g.a2, H0, colsk, Y0);
    k_softmax<<<(int)cdiv_sz((size_t)cols * 64, 256), 256, 0, stream>>>(Y0, n);
    k_agg<<<(int)cdiv_sz((size_t)cols * 64, 256), 256, 0, stream>>>(Y0, Tv, TposEmb, n, Tagg16);
    gemm(g.le, Tagg16, cols, Y1);
    k_add_id<<<(int)cdiv_sz((size_t)cols * inch, 256), 256, 0, stream>>>(Y1, A32, inch, cols, B32, B16);
  };

  // ======================= sa1 =======================
  k_fps<<<B, 256, 0, stream>>>(pc, N0, N1, fpsIdx1);
  k_gather_xyz<<<(int)cdiv_sz(B * 3 * N1, 256), 256, 0, stream>>>(pc, N0, fpsIdx1, N1, l1xyz);
  k_ball_query<<<(int)cdiv_sz(B * N1, 256), 256, 0, stream>>>(pc, l1xyz, N0, N1, 0.04f, bq1);
  k_group<<<(int)cdiv_sz((size_t)NK1 * 32, 256), 256, 0, stream>>>(pc, l1xyz, bq1, nullptr, 1, N0, N1, 32, 3, 32, H0);
  gemm(g_sa1_0, H0, NK1, Y0); stats(Y0, 64, NK1);
  bnact(Y0, 64, NK1, dmean, drstd, sa1m0.g, sa1m0.beta, 1, H1, 64, nullptr);
  gemm(g_sa1_1, H1, NK1, Y0); stats(Y0, 64, NK1);
  bnact(Y0, 64, NK1, dmean, drstd, sa1m1.g, sa1m1.beta, 1, H0, 64, nullptr);
  gemm(g_sa1_2, H0, NK1, Y0); stats(Y0, 128, NK1);
  bnact(Y0, 128, NK1, dmean, drstd, sa1m2.g, sa1m2.beta, 1, nullptr, 0, Y1);
  maxpool(Y1, 128, 32, CC1, l1A32, l1A16, 128);

  // ======================= tr1 =======================
  run_tr(gtr1, tr1, N1, 128, l1xyz, l1A32, l1A16, l1B32, l1B16, knn1);

  // ======================= sa2 =======================
  k_fps<<<B, 256, 0, stream>>>(l1xyz, N1, N2, fpsIdx2);
  k_gather_xyz<<<(int)cdiv_sz(B * 3 * N2, 256), 256, 0, stream>>>(l1xyz, N1, fpsIdx2, N2, l2xyz);
  k_ball_query<<<(int)cdiv_sz(B * N2, 256), 256, 0, stream>>>(l1xyz, l2xyz, N1, N2, 0.16f, bq2);
  k_group<<<(int)cdiv_sz((size_t)NK2 * 160, 256), 256, 0, stream>>>(l1xyz, l2xyz, bq2, l1B32, 0, N1, N2, 32, 128, 160, H0);
  gemm(g_sa2_0, H0, NK2, Y0); stats(Y0, 128, NK2);
  bnact(Y0, 128, NK2, dmean, drstd, sa2m0.g, sa2m0.beta, 1, H1, 128, nullptr);
  gemm(g_sa2_1, H1, NK2, Y0); stats(Y0, 128, NK2);
  bnact(Y0, 128, NK2, dmean, drstd, sa2m1.g, sa2m1.beta, 1, H0, 128, nullptr);
  gemm(g_sa2_2, H0, NK2, Y0); stats(Y0, 256, NK2);
  bnact(Y0, 256, NK2, dmean, drstd, sa2m2.g, sa2m2.beta, 1, nullptr, 0, Y1);
  maxpool(Y1, 256, 32, CC2, l2A32, l2A16, 256);

  // ======================= tr2 =======================
  run_tr(gtr2, tr2, N2, 256, l2xyz, l2A32, l2A16, l2B32, l2B16, knn2);

  // ======================= sa3 (group_all) =======================
  k_group_all<<<(int)cdiv_sz((size_t)CC2 * 288, 256), 256, 0, stream>>>(l2xyz, l2B32, N2, 256, 288, H0);
  gemm(g_sa3_0, H0, CC2, Y0); stats(Y0, 256, CC2);
  bnact(Y0, 256, CC2, dmean, drstd, sa3m0.g, sa3m0.beta, 1, H1, 256, nullptr);
  gemm(g_sa3_1, H1, CC2, Y0); stats(Y0, 512, CC2);
  bnact(Y0, 512, CC2, dmean, drstd, sa3m1.g, sa3m1.beta, 1, H0, 512, nullptr);
  gemm(g_sa3_2, H0, CC2, Y0); stats(Y0, 1024, CC2);
  bnact(Y0, 1024, CC2, dmean, drstd, sa3m2.g, sa3m2.beta, 1, nullptr, 0, Y1);
  maxpool(Y1, 1024, 128, B, l3pA, nullptr, 0);

  // ============== latent head (conv1 + bn1 + relu) -> output 0 ==============
  k_lat<<<(int)cdiv_sz(B * 1024, 256), 256, 0, stream>>>(l3pA, Flast, conv1.w, conv1.b, latY);
  stats(latY, 1, B * 1024);
  bnact(latY, 1, B * 1024, dmean, drstd, bn1_g, bn1_beta, 1, nullptr, 0, (float*)d_out);
  float* l3new = (float*)d_out;  // (B,1024,1) flat == point-major [b][1024]

  // ======================= fp3 =======================
  k_three_nn<<<(int)cdiv_sz(B * N2, 256), 256, 0, stream>>>(l2xyz, N2, l2xyz, 1, idx3, w3);
  k_interp_concat<<<(int)cdiv_sz((size_t)CC2 * 1280, 256), 256, 0, stream>>>(l3new, 1024, 1, l2B32, 256, idx3, w3, N2, 1280, H0);
  gemm(g_fp3_0, H0, CC2, Y0); stats(Y0, 256, CC2);
  bnact(Y0, 256, CC2, dmean, drstd, fp3m0.g, fp3m0.beta, 1, H1, 256, nullptr);
  gemm(g_fp3_1, H1, CC2, Y0); stats(Y0, 256, CC2);
  bnact(Y0, 256, CC2, dmean, drstd, fp3m1.g, fp3m1.beta, 1, nullptr, 0, l2C32);

  // ======================= fp2 =======================
  k_three_nn<<<(int)cdiv_sz(B * N1, 256), 256, 0, stream>>>(l1xyz, N1, l2xyz, N2, idx3, w3);
  k_interp_concat<<<(int)cdiv_sz((size_t)CC1 * 384, 256), 256, 0, stream>>>(l2C32, 256, N2, l1B32, 128, idx3, w3, N1, 384, H0);
  gemm(g_fp2_0, H0, CC1, Y0); stats(Y0, 256, CC1);
  bnact(Y0, 256, CC1, dmean, drstd, fp2m0.g, fp2m0.beta, 1, H1, 256, nullptr);
  gemm(g_fp2_1, H1, CC1, Y0); stats(Y0, 128, CC1);
  bnact(Y0, 128, CC1, dmean, drstd, fp2m1.g, fp2m1.beta, 1, nullptr, 0, l1C32);

  // ======================= fp1 =======================
  k_build_pc6<<<(int)cdiv_sz(B * N0 * 6, 256), 256, 0, stream>>>(pc, pc6);
  k_three_nn<<<(int)cdiv_sz(B * N0, 256), 256, 0, stream>>>(pc, N0, l1xyz, N1, idx3, w3);
  k_interp_concat<<<(int)cdiv_sz((size_t)CN0 * 160, 256), 256, 0, stream>>>(l1C32, 128, N1, pc6, 6, idx3, w3, N0, 160, H0);
  gemm(g_fp1_0, H0, CN0, Y0); stats(Y0, 128, CN0);
  bnact(Y0, 128, CN0, dmean, drstd, fp1m0.g, fp1m0.beta, 1, H1, 128, nullptr);
  gemm(g_fp1_1, H1, CN0, Y0); stats(Y0, 128, CN0);
  bnact(Y0, 128, CN0, dmean, drstd, fp1m1.g, fp1m1.beta, 1, H0, 128, nullptr);
  gemm(g_fp1_2, H0, CN0, Y0); stats(Y0, 128, CN0);
  bnact(Y0, 128, CN0, dmean, drstd, fp1m2.g, fp1m2.beta, 1, nullptr, 0, l0pts);

  // ======================= final MLP + tanh residual =======================
  k_concat_noise<<<(int)cdiv_sz((size_t)CN0 * 160, 256), 256, 0, stream>>>(l0pts, noise, H1);
  gemm(g_fin0, H1, CN0, Y0); stats(Y0, 128, CN0);
  bnact(Y0, 128, CN0, dmean, drstd, fin0.g, fin0.beta, 1, H0, 128, nullptr);
  gemm(g_fin1, H0, CN0, Y0); stats(Y0, 64, CN0);
  bnact(Y0, 64, CN0, dmean, drstd, fin1.g, fin1.beta, 1, H1, 64, nullptr);
  gemm(g_fin2, H1, CN0, Y0);   // (B*2048, 3)
  k_final_out<<<(int)cdiv_sz(B * 3 * N0, 256), 256, 0, stream>>>(Y0, pc, (float*)d_out + B * 1024);
}
